// MultiHeadAttention_78365973282907
// MI455X (gfx1250) — compile-verified, matrix operand order checked by reading, not run on hardware
//
#include <hip/hip_runtime.h>

#define B_   2
#define S_   2048
#define HID_ 2048
#define H_   32
#define KVH_ 8
#define D_   64
#define G_   4

typedef __attribute__((ext_vector_type(16))) __bf16          v16bf;
typedef __attribute__((ext_vector_type(8)))  float           v8f;
typedef __attribute__((ext_vector_type(8)))  unsigned short  ushort8;
typedef unsigned short u16;

union FragA { v16bf v; ushort8 h[2]; };

__device__ __forceinline__ u16 f2bf(float f) {
  unsigned u = __float_as_uint(f);
  u += 0x7FFFu + ((u >> 16) & 1u);          // round-to-nearest-even
  return (u16)(u >> 16);
}
__device__ __forceinline__ float bf2f(u16 h) {
  return __uint_as_float(((unsigned)h) << 16);
}

// ---------------------------------------------------------------------------
// CDNA5 async global->LDS copy (GVS mode: SGPR base + per-lane byte offset).
// vdst VGPR carries the LDS byte address; tracked with ASYNCcnt.
// ---------------------------------------------------------------------------
__device__ __forceinline__ void async_ld_b128(unsigned lds_off, const void* base,
                                              unsigned byte_off) {
  asm volatile("global_load_async_to_lds_b128 %0, %1, %2"
               :: "v"(lds_off), "v"(byte_off), "s"(base) : "memory");
}
__device__ __forceinline__ void async_wait0() {
  asm volatile("s_wait_asynccnt 0" ::: "memory");
}

// ---------------------------------------------------------------------------
// Generic C = A (MxK) * W^T (NxK), fp32 accumulate via bf16 WMMA.
// Block: 256 threads (8 waves), tile 128(M) x 128(N), K stepped by 32 via LDS.
// Each wave owns a 32x64 sub-tile: 2 A-frags x 4 B-frags -> 8 WMMA / K-step.
// A is staged via async-LDS copy when already bf16, else VALU-converted.
// ---------------------------------------------------------------------------
template<bool A_BF16, bool OUT_BF16>
__global__ __launch_bounds__(256)
void gemm_nt_kernel(const void* __restrict__ Av, const float* __restrict__ W,
                    void* __restrict__ Cv, int M, int N, int K)
{
  __shared__ __align__(16) u16 As[128][32];
  __shared__ __align__(16) u16 Ws[128][32];

  const int m0  = blockIdx.y * 128;
  const int n0  = blockIdx.x * 128;
  const int tid = threadIdx.x;
  const int lane = tid & 31;
  const int wav  = tid >> 5;
  const int wm   = wav >> 1;        // 0..3 : 32-row strip
  const int wn   = wav & 1;         // 0..1 : 64-col strip
  const int lh   = lane & 15;
  const int khi  = lane >> 4;
  const int kb   = khi * 8;

  const float* Af = (const float*)Av;
  const u16*   Ah = (const u16*)Av;

  const v8f vzero = {0.f,0.f,0.f,0.f,0.f,0.f,0.f,0.f};
  v8f acc[2][4];
  #pragma unroll
  for (int a = 0; a < 2; ++a)
    #pragma unroll
    for (int t = 0; t < 4; ++t) acc[a][t] = vzero;

  for (int k0 = 0; k0 < K; k0 += 32) {
    __syncthreads();
    // Stage A(128x32) and W(128x32) as bf16: 512 chunks of 8 elems each side.
    #pragma unroll
    for (int i = 0; i < 2; ++i) {
      const int c  = tid * 2 + i;
      const int r  = c >> 2;
      const int c0 = (c & 3) * 8;
      if (A_BF16) {
        async_ld_b128((unsigned)(size_t)&As[r][c0], Ah,
                      (unsigned)(((size_t)(m0 + r) * K + k0 + c0) * 2));
      } else {
        const float* ap = Af + (size_t)(m0 + r) * K + k0 + c0;
        float4 x0 = *(const float4*)(ap);
        float4 x1 = *(const float4*)(ap + 4);
        ushort8 sv;
        sv[0]=f2bf(x0.x); sv[1]=f2bf(x0.y); sv[2]=f2bf(x0.z); sv[3]=f2bf(x0.w);
        sv[4]=f2bf(x1.x); sv[5]=f2bf(x1.y); sv[6]=f2bf(x1.z); sv[7]=f2bf(x1.w);
        *(ushort8*)&As[r][c0] = sv;
      }
      const float* wp = W + (size_t)(n0 + r) * K + k0 + c0;
      if (k0 + 32 < K) __builtin_prefetch(wp + 32, 0, 1);   // global_prefetch_b8
      float4 w0 = *(const float4*)(wp);
      float4 w1 = *(const float4*)(wp + 4);
      ushort8 wv;
      wv[0]=f2bf(w0.x); wv[1]=f2bf(w0.y); wv[2]=f2bf(w0.z); wv[3]=f2bf(w0.w);
      wv[4]=f2bf(w1.x); wv[5]=f2bf(w1.y); wv[6]=f2bf(w1.z); wv[7]=f2bf(w1.w);
      *(ushort8*)&Ws[r][c0] = wv;
    }
    if (A_BF16) async_wait0();
    __syncthreads();

    FragA fa[2];
    #pragma unroll
    for (int a = 0; a < 2; ++a) {
      const int ar = 32 * wm + 16 * a + lh;
      fa[a].h[0] = *(const ushort8*)&As[ar][kb];
      fa[a].h[1] = *(const ushort8*)&As[ar][kb + 16];
    }
    #pragma unroll
    for (int t = 0; t < 4; ++t) {
      FragA fb;
      const int br = 64 * wn + 16 * t + lh;
      fb.h[0] = *(const ushort8*)&Ws[br][kb];
      fb.h[1] = *(const ushort8*)&Ws[br][kb + 16];
      #pragma unroll
      for (int a = 0; a < 2; ++a)
        acc[a][t] = __builtin_amdgcn_wmma_f32_16x16x32_bf16(
                        false, fa[a].v, false, fb.v, (short)0, acc[a][t],
                        false, false);
    }
  }

  #pragma unroll
  for (int a = 0; a < 2; ++a) {
    #pragma unroll
    for (int t = 0; t < 4; ++t) {
      #pragma unroll
      for (int r = 0; r < 8; ++r) {
        const int gm = m0 + 32 * wm + 16 * a + r + khi * 8;
        const int gn = n0 + 64 * wn + 16 * t + lh;
        const float v = acc[a][t][r];
        if (OUT_BF16) ((u16*)Cv)[(size_t)gm * N + gn] = f2bf(v);
        else          ((float*)Cv)[(size_t)gm * N + gn] = v;
      }
    }
  }
}

// ---------------------------------------------------------------------------
// RoPE + repack: qproj(B,S,H*D) -> Qp(B,H,S,D); kproj -> Kp(B,KVH,S,D);
// vproj -> Vt(B,KVH,D,S)  (all bf16).
// ---------------------------------------------------------------------------
__global__ void rope_pack_kernel(const u16* __restrict__ qproj,
                                 const u16* __restrict__ kproj,
                                 const u16* __restrict__ vproj,
                                 const float* __restrict__ cosb,
                                 const float* __restrict__ sinb,
                                 u16* __restrict__ Qp, u16* __restrict__ Kp,
                                 u16* __restrict__ Vtb)
{
  const int idx = blockIdx.x * blockDim.x + threadIdx.x;   // over B*S*H*D = 2^23
  if (idx >= B_ * S_ * H_ * D_) return;
  const int d = idx & 63;
  const int h = (idx >> 6) & 31;
  const int s = (idx >> 11) & 2047;
  const int b = idx >> 22;

  const float c  = cosb[((size_t)b * S_ + s) * D_ + d];
  const float sn = sinb[((size_t)b * S_ + s) * D_ + d];
  const int   pd = d ^ 32;                   // rotate_half partner
  const float sgn = (d < 32) ? -1.f : 1.f;

  {
    const u16* qrow = qproj + ((size_t)b * S_ + s) * (H_ * D_) + h * D_;
    const float q  = bf2f(qrow[d]);
    const float qp = bf2f(qrow[pd]);
    Qp[(((size_t)b * H_ + h) * S_ + s) * D_ + d] = f2bf(q * c + sgn * qp * sn);
  }
  if (h < KVH_) {
    const u16* krow = kproj + ((size_t)b * S_ + s) * (KVH_ * D_) + h * D_;
    const float k  = bf2f(krow[d]);
    const float kp = bf2f(krow[pd]);
    Kp[(((size_t)b * KVH_ + h) * S_ + s) * D_ + d] = f2bf(k * c + sgn * kp * sn);
    const u16* vrow = vproj + ((size_t)b * S_ + s) * (KVH_ * D_) + h * D_;
    Vtb[(((size_t)b * KVH_ + h) * D_ + d) * S_ + s] = vrow[d];
  }
}

// ---------------------------------------------------------------------------
// Flash-style attention. Block = (b,h, 64 q-rows), 4 waves, each wave owns a
// 16-row strip. K / V^T tiles staged with async global->LDS copies (ASYNCcnt).
// Q·K^T via WMMA, online softmax in C-fragment layout (shfl reductions across
// 16-lane halves), P re-shaped via per-wave LDS strip, P·V via WMMA.
// ---------------------------------------------------------------------------
__global__ __launch_bounds__(128)
void attn_kernel(const u16* __restrict__ Qp, const u16* __restrict__ Kp,
                 const u16* __restrict__ Vtb,
                 const unsigned char* __restrict__ mask,
                 u16* __restrict__ Op)
{
  __shared__ __align__(16) u16 Ks[64][64];        // [key][d]
  __shared__ __align__(16) u16 Vs[64][64];        // [d][key]   (V^T tile)
  __shared__ __align__(16) u16 Ps[4][16][64];     // per-wave P strip

  const int tid  = threadIdx.x;
  const int lane = tid & 31;
  const int wav  = tid >> 5;
  const int lh   = lane & 15;
  const int khi  = lane >> 4;
  const int kb   = khi * 8;

  const int bh  = blockIdx.y;
  const int b   = bh / H_;
  const int h   = bh % H_;
  const int kvh = h / G_;
  const int s0  = blockIdx.x * 64;

  // Preload Q fragments (row = s0 + 16*wav + lh).
  FragA qa[2];
  {
    const u16* qbase = Qp + (((size_t)b * H_ + h) * S_ + (s0 + 16 * wav + lh)) * D_;
    #pragma unroll
    for (int dk = 0; dk < 2; ++dk) {
      qa[dk].h[0] = *(const ushort8*)(qbase + dk * 32 + kb);
      qa[dk].h[1] = *(const ushort8*)(qbase + dk * 32 + kb + 16);
    }
  }

  const v8f vzero = {0.f,0.f,0.f,0.f,0.f,0.f,0.f,0.f};
  float m_run[8], l_run[8];
  v8f o[4];
  #pragma unroll
  for (int r = 0; r < 8; ++r) { m_run[r] = -1e30f; l_run[r] = 0.f; }
  #pragma unroll
  for (int t = 0; t < 4; ++t) o[t] = vzero;

  const u16* kbase = Kp  + ((size_t)b * KVH_ + kvh) * S_ * D_;
  const u16* vbase = Vtb + ((size_t)b * KVH_ + kvh) * D_ * S_;
  const unsigned char* mrow = mask + (size_t)b * S_;
  const float LOG2E = 1.4426950408889634f;

  for (int j0 = 0; j0 < S_; j0 += 64) {
    __syncthreads();
    // Async stage: K tile (8KB) + V^T tile (8KB); 4x b128 each per thread.
    #pragma unroll
    for (int i = 0; i < 4; ++i) {
      const int c  = tid * 4 + i;          // 512 chunks
      const int r  = c >> 3;
      const int c0 = (c & 7) * 8;
      async_ld_b128((unsigned)(size_t)&Ks[r][c0], kbase,
                    (unsigned)(((j0 + r) * D_ + c0) * 2));
      async_ld_b128((unsigned)(size_t)&Vs[r][c0], vbase,
                    (unsigned)((r * S_ + j0 + c0) * 2));
    }
    async_wait0();
    __syncthreads();

    // scores = Q (16xD) . K^T -> 4 tiles of 16x16
    v8f sc[4];
    #pragma unroll
    for (int t = 0; t < 4; ++t) {
      sc[t] = vzero;
      #pragma unroll
      for (int dk = 0; dk < 2; ++dk) {
        FragA fb;
        fb.h[0] = *(const ushort8*)&Ks[16 * t + lh][dk * 32 + kb];
        fb.h[1] = *(const ushort8*)&Ks[16 * t + lh][dk * 32 + kb + 16];
        sc[t] = __builtin_amdgcn_wmma_f32_16x16x32_bf16(
                    false, qa[dk].v, false, fb.v, (short)0, sc[t], false, false);
      }
    }

    // scale + mask (column j = j0 + 16t + lh, same for all r in a vgpr)
    #pragma unroll
    for (int t = 0; t < 4; ++t) {
      const bool mk = mrow[j0 + 16 * t + lh] != 0;
      #pragma unroll
      for (int r = 0; r < 8; ++r)
        sc[t][r] = mk ? sc[t][r] * 0.125f : -1e30f;
    }

    // online softmax: row stats live in matching (r, lane-half) layout
    float mnew[8], scl[8], rs[8];
    #pragma unroll
    for (int r = 0; r < 8; ++r) {
      float mx = fmaxf(fmaxf(sc[0][r], sc[1][r]), fmaxf(sc[2][r], sc[3][r]));
      #pragma unroll
      for (int off = 1; off < 16; off <<= 1)
        mx = fmaxf(mx, __shfl_xor(mx, off, 32));
      mnew[r] = fmaxf(m_run[r], mx);
      scl[r]  = exp2f((m_run[r] - mnew[r]) * LOG2E);
      rs[r]   = 0.f;
    }
    #pragma unroll
    for (int t = 0; t < 4; ++t) {
      #pragma unroll
      for (int r = 0; r < 8; ++r) {
        const float p = exp2f((sc[t][r] - mnew[r]) * LOG2E);
        rs[r] += p;
        Ps[wav][r + khi * 8][16 * t + lh] = f2bf(p);
      }
    }
    #pragma unroll
    for (int r = 0; r < 8; ++r) {
      float s = rs[r];
      #pragma unroll
      for (int off = 1; off < 16; off <<= 1)
        s += __shfl_xor(s, off, 32);
      l_run[r] = l_run[r] * scl[r] + s;
      m_run[r] = mnew[r];
    }
    #pragma unroll
    for (int t = 0; t < 4; ++t)
      #pragma unroll
      for (int r = 0; r < 8; ++r) o[t][r] *= scl[r];

    // O += P (16x64) . V (64xD)
    #pragma unroll
    for (int ks = 0; ks < 2; ++ks) {
      FragA pa;
      pa.h[0] = *(const ushort8*)&Ps[wav][lh][ks * 32 + kb];
      pa.h[1] = *(const ushort8*)&Ps[wav][lh][ks * 32 + kb + 16];
      #pragma unroll
      for (int t = 0; t < 4; ++t) {
        FragA vb;
        vb.h[0] = *(const ushort8*)&Vs[16 * t + lh][ks * 32 + kb];
        vb.h[1] = *(const ushort8*)&Vs[16 * t + lh][ks * 32 + kb + 16];
        o[t] = __builtin_amdgcn_wmma_f32_16x16x32_bf16(
                   false, pa.v, false, vb.v, (short)0, o[t], false, false);
      }
    }
  }

  // epilogue: O / l, write to (B,S,H*D) bf16
  #pragma unroll
  for (int t = 0; t < 4; ++t) {
    #pragma unroll
    for (int r = 0; r < 8; ++r) {
      const int srow = s0 + 16 * wav + r + khi * 8;
      const int col  = h * D_ + 16 * t + lh;
      Op[((size_t)b * S_ + srow) * (H_ * D_) + col] = f2bf(o[t][r] / l_run[r]);
    }
  }
}

// ---------------------------------------------------------------------------
extern "C" void kernel_launch(void* const* d_in, const int* in_sizes, int n_in,
                              void* d_out, int out_size, void* d_ws, size_t ws_size,
                              hipStream_t stream)
{
  (void)in_sizes; (void)n_in; (void)out_size; (void)ws_size;
  const float*         hidden = (const float*)d_in[0];
  const float*         cosb   = (const float*)d_in[1];
  const float*         sinb   = (const float*)d_in[2];
  const unsigned char* maskp  = (const unsigned char*)d_in[3];
  const float*         Wq     = (const float*)d_in[4];
  const float*         Wk     = (const float*)d_in[5];
  const float*         Wv     = (const float*)d_in[6];
  const float*         Wo     = (const float*)d_in[7];

  char* ws = (char*)d_ws;
  size_t off = 0;
  auto wsalloc = [&](size_t bytes) -> void* {
    void* p = ws + off;
    off += (bytes + 255) & ~(size_t)255;
    return p;
  };
  const size_t qsz = (size_t)B_ * S_ * H_   * D_ * 2;   // bf16
  const size_t ksz = (size_t)B_ * S_ * KVH_ * D_ * 2;
  u16* qproj = (u16*)wsalloc(qsz);
  u16* kproj = (u16*)wsalloc(ksz);
  u16* vproj = (u16*)wsalloc(ksz);
  u16* Qp    = (u16*)wsalloc(qsz);
  u16* Kp    = (u16*)wsalloc(ksz);
  u16* Vtb   = (u16*)wsalloc(ksz);
  u16* Opb   = (u16*)wsalloc(qsz);

  const int M = B_ * S_;
  dim3 blk(256);

  gemm_nt_kernel<false, true><<<dim3(HID_ / 128, M / 128), blk, 0, stream>>>(
      hidden, Wq, qproj, M, HID_, HID_);
  gemm_nt_kernel<false, true><<<dim3((KVH_ * D_) / 128, M / 128), blk, 0, stream>>>(
      hidden, Wk, kproj, M, KVH_ * D_, HID_);
  gemm_nt_kernel<false, true><<<dim3((KVH_ * D_) / 128, M / 128), blk, 0, stream>>>(
      hidden, Wv, vproj, M, KVH_ * D_, HID_);

  const int n_rope = B_ * S_ * H_ * D_;
  rope_pack_kernel<<<(n_rope + 255) / 256, 256, 0, stream>>>(
      qproj, kproj, vproj, cosb, sinb, Qp, Kp, Vtb);

  attn_kernel<<<dim3(S_ / 64, B_ * H_), dim3(128), 0, stream>>>(
      Qp, Kp, Vtb, maskp, Opb);

  gemm_nt_kernel<true, false><<<dim3(HID_ / 128, M / 128), blk, 0, stream>>>(
      Opb, Wo, d_out, M, HID_, HID_);
}
